// SingleHeadAttention_23811298689314
// MI455X (gfx1250) — matrix operand order repaired and checked by reading, not compiled, not run on hardware
//
#include <hip/hip_runtime.h>
#include <hip/hip_bf16.h>

// ---------------------------------------------------------------------------
// Single-head causal attention for MI455X (gfx1250, wave32, WMMA).
// Pipeline: f32->f16 convert | QKV proj GEMM | S=Q*K^T (causal) | softmax | O=P*V
// GEMMs: v_wmma_f32_16x16x32_f16, B-operand tile double-buffered in LDS via
// global_load_async_to_lds_b128 (ASYNCcnt) when available.
// ---------------------------------------------------------------------------

typedef __attribute__((ext_vector_type(16))) _Float16 v16h;
typedef __attribute__((ext_vector_type(8)))  _Float16 v8h;
typedef __attribute__((ext_vector_type(8)))  float    v8f;

typedef __attribute__((address_space(3))) _Float16 lf16;
typedef __attribute__((address_space(3))) v8h      lv8h;

// gcc-style vector to match the async-LDS builtin's parameter type exactly
typedef int v4i_vs __attribute__((vector_size(16)));

#define CAT16(lo, hi) __builtin_shufflevector((lo), (hi), 0,1,2,3,4,5,6,7,8,9,10,11,12,13,14,15)

#if __has_builtin(__builtin_amdgcn_global_load_async_to_lds_b128)
#define HAVE_ASYNC_LDS 1
#else
#define HAVE_ASYNC_LDS 0
#endif

// LDS tile geometry: 64 rows x 32 f16, row stride padded to 40 f16 (80 B) so
// the 16 b128 start banks (20n mod 64, all distinct multiples of 4) are
// conflict-free. Double buffered: 2 x 5120 B = 10 KB.
#define LDS_STRIDE 40
#define LDS_TILE   (64 * LDS_STRIDE)

__device__ __forceinline__ v8f wmma_f16(v16h a, v16h b, v8f c) {
  // (neg_a, A, neg_b, B, c_mod, C, reuse_a, reuse_b)
  return __builtin_amdgcn_wmma_f32_16x16x32_f16(false, a, false, b, (short)0, c, false, false);
}

__device__ __forceinline__ void zero_acc(v8f* acc) {
#pragma unroll
  for (int n = 0; n < 4; ++n)
#pragma unroll
    for (int i = 0; i < 8; ++i) acc[n][i] = 0.0f;
}

// A-fragment (16x32 f16, M x K): lane l (r=l&15, h=l>>4) holds row M=r,
// K in [k0+16h, k0+16h+16) contiguous  -> two 16B loads.
__device__ __forceinline__ v16h load_afrag(const _Float16* rowPtr, int k0, int half) {
  v8h lo = *(const v8h*)(rowPtr + k0 + half * 16);
  v8h hi = *(const v8h*)(rowPtr + k0 + half * 16 + 8);
  return CAT16(lo, hi);
}

// B-fragment (32x16 f16, K x N) from the LDS-staged row-major N x K tile:
// lane l (n=l&15, h=l>>4) holds column N=n, K contiguous per lane-half.
__device__ __forceinline__ v16h lds_bfrag(const lf16* p) {
  v8h lo = *(const lv8h*)(p);
  v8h hi = *(const lv8h*)(p + 8);
  return CAT16(lo, hi);
}

// One 16-byte global -> LDS copy per thread (256 threads cover a 64x32 tile).
__device__ __forceinline__ void stage16(const _Float16* g, lf16* l) {
#if HAVE_ASYNC_LDS
  __builtin_amdgcn_global_load_async_to_lds_b128(
      (__attribute__((address_space(1))) v4i_vs*)g,
      (__attribute__((address_space(3))) v4i_vs*)l, 0, 0);
#else
  *(lv8h*)l = *(const v8h*)g;
#endif
}

template <int N>
__device__ __forceinline__ void wait_async() {
#if HAVE_ASYNC_LDS
#if __has_builtin(__builtin_amdgcn_s_wait_asynccnt)
  __builtin_amdgcn_s_wait_asynccnt(N);
#else
  if constexpr (N == 0) asm volatile("s_wait_asynccnt 0x0" ::: "memory");
  else                  asm volatile("s_wait_asynccnt 0x1" ::: "memory");
#endif
#endif
}

// ---------------------------------------------------------------------------
// K0: f32 -> f16 convert
// ---------------------------------------------------------------------------
__global__ __launch_bounds__(256) void cvt_f32_to_f16(const float* __restrict__ src,
                                                      _Float16* __restrict__ dst, int n) {
  int i = blockIdx.x * 256 + threadIdx.x;
  if (i < n) dst[i] = (_Float16)src[i];
}

// ---------------------------------------------------------------------------
// K1: QKV projection.  C[m, a] = sum_e Xh[m, e] * W[a, e]
//   z = 0 -> Q (scaled by 1/sqrt(A) = 1/32), z = 1 -> K  (row-major (B*T) x A)
//   z = 2 -> V, stored transposed per batch: Vt[(b*A + a)*T + t]
// Block: 8 waves; wave tile 16(M) x 64(N); W tile (shared by all waves)
// double-buffered in LDS via async loads.
// ---------------------------------------------------------------------------
__global__ __launch_bounds__(256) void qkv_proj(const _Float16* __restrict__ Xh,
                                                const _Float16* __restrict__ Wh,
                                                _Float16* __restrict__ Qb,
                                                _Float16* __restrict__ Kb,
                                                _Float16* __restrict__ Vb) {
  constexpr int T = 4096, E = 1024, A = 1024;
  __shared__ _Float16 wtile[2 * LDS_TILE];

  const int tid = threadIdx.x;
  const int wid = tid >> 5, lane = tid & 31;
  const int r = lane & 15, half = lane >> 4;
  const int z  = blockIdx.z;
  const int m0 = blockIdx.x * 128 + wid * 16;
  const int c0 = blockIdx.y * 64;

  lf16* lt = (lf16*)&wtile[0];
  const _Float16* Wtile = Wh + (size_t)z * A * E + (size_t)c0 * E;  // 64 rows, stride E
  const _Float16* arow  = Xh + (size_t)(m0 + r) * E;

  const int trow = tid >> 2, tc = (tid & 3) << 3;
  auto stage_tile = [&](int buf, int k0) {
    stage16(Wtile + (size_t)trow * E + k0 + tc, lt + buf * LDS_TILE + trow * LDS_STRIDE + tc);
  };

  v8f acc[4];
  zero_acc(acc);

  constexpr int nIter = E / 32;
  stage_tile(0, 0);
  for (int it = 0; it < nIter; ++it) {
    const int k0 = it * 32, buf = it & 1;
    if (it + 1 < nIter) { stage_tile(buf ^ 1, k0 + 32); wait_async<1>(); }
    else                { wait_async<0>(); }
    __syncthreads();

    v16h af = load_afrag(arow, k0, half);
    const lf16* lb = lt + buf * LDS_TILE;
#pragma unroll
    for (int nt = 0; nt < 4; ++nt) {
      v16h bf = lds_bfrag(lb + (nt * 16 + r) * LDS_STRIDE + half * 16);
      acc[nt] = wmma_f16(af, bf, acc[nt]);
    }
    __syncthreads();
  }

  // C/D layout: M = vi + 8*half, N = r
  if (z < 2) {
    _Float16* D = (z == 0) ? Qb : Kb;
    const float scale = (z == 0) ? 0.03125f : 1.0f;  // 1/sqrt(1024)
#pragma unroll
    for (int nt = 0; nt < 4; ++nt)
#pragma unroll
      for (int vi = 0; vi < 8; ++vi) {
        int m = m0 + vi + 8 * half;
        int c = c0 + nt * 16 + r;
        D[(size_t)m * A + c] = (_Float16)(acc[nt][vi] * scale);
      }
  } else {
#pragma unroll
    for (int nt = 0; nt < 4; ++nt)
#pragma unroll
      for (int vi = 0; vi < 8; ++vi) {
        int m = m0 + vi + 8 * half;
        int c = c0 + nt * 16 + r;
        int b = m >> 12, t = m & (T - 1);
        Vb[((size_t)b * A + c) * T + t] = (_Float16)acc[nt][vi];
      }
  }
}

// ---------------------------------------------------------------------------
// K2: S[b, q, k] = Q[b,q,:] . K[b,k,:]   (Q pre-scaled), causal.
// Wave tile 16 q x 64 k; K tile (64 keys x 32 a, shared by all waves)
// double-buffered in LDS. Key tiles past the 32-aligned causal boundary
// B32 = ((q0+15)/32 + 1)*32 are skipped (compute guard, barriers uniform).
// ---------------------------------------------------------------------------
__global__ __launch_bounds__(256) void qk_scores(const _Float16* __restrict__ Qb,
                                                 const _Float16* __restrict__ Kb,
                                                 _Float16* __restrict__ Sb) {
  constexpr int T = 4096, A = 1024;
  if ((int)(blockIdx.y * 64) >= (int)(blockIdx.x * 128 + 128)) return;  // block-uniform skip

  __shared__ _Float16 ktile[2 * LDS_TILE];

  const int tid = threadIdx.x;
  const int wid = tid >> 5, lane = tid & 31;
  const int r = lane & 15, half = lane >> 4;
  const int b  = blockIdx.z;
  const int q0 = blockIdx.x * 128 + wid * 16;
  const int c0 = blockIdx.y * 64;
  const int B32 = (((q0 + 15) >> 5) + 1) << 5;  // causal pad boundary for this wave

  bool act[4];
#pragma unroll
  for (int nt = 0; nt < 4; ++nt) act[nt] = (c0 + nt * 16) < B32;

  lf16* lt = (lf16*)&ktile[0];
  const _Float16* Ktile = Kb + (size_t)(b * T + c0) * A;  // 64 key rows, stride A
  const _Float16* qrow  = Qb + (size_t)(b * T + q0 + r) * A;

  const int trow = tid >> 2, tc = (tid & 3) << 3;
  auto stage_tile = [&](int buf, int k0) {
    stage16(Ktile + (size_t)trow * A + k0 + tc, lt + buf * LDS_TILE + trow * LDS_STRIDE + tc);
  };

  v8f acc[4];
  zero_acc(acc);

  constexpr int nIter = A / 32;
  stage_tile(0, 0);
  for (int it = 0; it < nIter; ++it) {
    const int k0 = it * 32, buf = it & 1;
    if (it + 1 < nIter) { stage_tile(buf ^ 1, k0 + 32); wait_async<1>(); }
    else                { wait_async<0>(); }
    __syncthreads();

    if (act[0]) {
      v16h af = load_afrag(qrow, k0, half);
      const lf16* lb = lt + buf * LDS_TILE;
#pragma unroll
      for (int nt = 0; nt < 4; ++nt) {
        if (act[nt]) {
          v16h bf = lds_bfrag(lb + (nt * 16 + r) * LDS_STRIDE + half * 16);
          acc[nt] = wmma_f16(af, bf, acc[nt]);
        }
      }
    }
    __syncthreads();
  }

  _Float16* S = Sb + (size_t)b * T * T;
#pragma unroll
  for (int nt = 0; nt < 4; ++nt) {
    if (!act[nt]) continue;
#pragma unroll
    for (int vi = 0; vi < 8; ++vi) {
      int q  = q0 + vi + 8 * half;
      int kc = c0 + nt * 16 + r;
      float v = acc[nt][vi];
      if (kc > q) v = -INFINITY;  // causal mask
      S[(size_t)q * T + kc] = (_Float16)v;
    }
  }
}

// ---------------------------------------------------------------------------
// K3: in-place row softmax over S[0 .. B32) for each (b, q) row.
// Row staged in LDS; wave32 shuffle + LDS reductions; exp via v_exp_f32.
// Masked (-inf) entries become exact 0 -> P is zero-padded to the 32 boundary.
// ---------------------------------------------------------------------------
__global__ __launch_bounds__(256) void softmax_rows(_Float16* __restrict__ Sb) {
  constexpr int T = 4096;
  __shared__ float srow[4096];
  __shared__ float red[8];
  __shared__ float bc;

  const int row = blockIdx.x;
  const int b = row >> 12, q = row & (T - 1);
  const int len = ((q >> 5) + 1) << 5;
  _Float16* S = Sb + (size_t)b * T * T + (size_t)q * T;

  const int tid = threadIdx.x;
  const int wid = tid >> 5, lane = tid & 31;

  float m = -INFINITY;
  for (int k = tid; k < len; k += 256) {
    float s = (float)S[k];
    srow[k] = s;
    m = fmaxf(m, s);
  }
#pragma unroll
  for (int off = 16; off > 0; off >>= 1) m = fmaxf(m, __shfl_xor(m, off));
  if (lane == 0) red[wid] = m;
  __syncthreads();
  if (tid == 0) {
    float mm = red[0];
#pragma unroll
    for (int i = 1; i < 8; ++i) mm = fmaxf(mm, red[i]);
    bc = mm;
  }
  __syncthreads();
  const float M = bc;

  float p = 0.0f;
  for (int k = tid; k < len; k += 256) p += __expf(srow[k] - M);
#pragma unroll
  for (int off = 16; off > 0; off >>= 1) p += __shfl_xor(p, off);
  if (lane == 0) red[wid] = p;
  __syncthreads();
  if (tid == 0) {
    float ss = red[0];
#pragma unroll
    for (int i = 1; i < 8; ++i) ss += red[i];
    bc = 1.0f / ss;
  }
  __syncthreads();
  const float inv = bc;

  for (int k = tid; k < len; k += 256)
    S[k] = (_Float16)(__expf(srow[k] - M) * inv);
}

// ---------------------------------------------------------------------------
// K4: O[b, q, a] = sum_k P[b,q,k] * V[b,k,a].  Wave tile 16 q x 64 a.
// Vt tile (64 a-rows x 32 keys, shared by all waves) double-buffered in LDS.
// Block iterates to the block-max causal bound; waves guard compute at their
// own boundary (P zero-padded there), barriers stay uniform.
// ---------------------------------------------------------------------------
__global__ __launch_bounds__(256) void attn_out(const _Float16* __restrict__ Pb,
                                                const _Float16* __restrict__ Vb,
                                                float* __restrict__ Out) {
  constexpr int T = 4096, A = 1024;
  __shared__ _Float16 vtile[2 * LDS_TILE];

  const int tid = threadIdx.x;
  const int wid = tid >> 5, lane = tid & 31;
  const int r = lane & 15, half = lane >> 4;
  const int b  = blockIdx.z;
  const int q0 = blockIdx.x * 128 + wid * 16;
  const int c0 = blockIdx.y * 64;
  const int kendW   = (((q0 + 15) >> 5) + 1) << 5;   // this wave's causal bound
  const int kendMax = blockIdx.x * 128 + 128;        // block-uniform bound

  lf16* lt = (lf16*)&vtile[0];
  const _Float16* Vtile = Vb + ((size_t)b * A + c0) * T;  // 64 a-rows, stride T
  const _Float16* prow  = Pb + (size_t)b * T * T + (size_t)(q0 + r) * T;

  const int trow = tid >> 2, tc = (tid & 3) << 3;
  auto stage_tile = [&](int buf, int k0) {
    stage16(Vtile + (size_t)trow * T + k0 + tc, lt + buf * LDS_TILE + trow * LDS_STRIDE + tc);
  };

  v8f acc[4];
  zero_acc(acc);

  const int nIter = kendMax / 32;
  stage_tile(0, 0);
  for (int it = 0; it < nIter; ++it) {
    const int k0 = it * 32, buf = it & 1;
    if (it + 1 < nIter) { stage_tile(buf ^ 1, k0 + 32); wait_async<1>(); }
    else                { wait_async<0>(); }
    __syncthreads();

    if (k0 < kendW) {
      v16h af = load_afrag(prow, k0, half);
      const lf16* lb = lt + buf * LDS_TILE;
#pragma unroll
      for (int nt = 0; nt < 4; ++nt) {
        v16h bf = lds_bfrag(lb + (nt * 16 + r) * LDS_STRIDE + half * 16);
        acc[nt] = wmma_f16(af, bf, acc[nt]);
      }
    }
    __syncthreads();
  }

#pragma unroll
  for (int nt = 0; nt < 4; ++nt)
#pragma unroll
    for (int vi = 0; vi < 8; ++vi) {
      int q = q0 + vi + 8 * half;
      int a = c0 + nt * 16 + r;
      Out[((size_t)b * T + q) * A + a] = acc[nt][vi];
    }
}

// ---------------------------------------------------------------------------
// Host launcher
// ---------------------------------------------------------------------------
extern "C" void kernel_launch(void* const* d_in, const int* in_sizes, int n_in,
                              void* d_out, int out_size, void* d_ws, size_t ws_size,
                              hipStream_t stream) {
  (void)in_sizes; (void)n_in; (void)out_size; (void)ws_size;
  constexpr int B = 4, T = 4096, E = 1024, A = 1024;
  constexpr size_t BT = (size_t)B * T;

  const float* x  = (const float*)d_in[0];
  const float* Wq = (const float*)d_in[1];
  const float* Wk = (const float*)d_in[2];
  const float* Wv = (const float*)d_in[3];

  // Workspace carve-up (f16 elements)
  char* ws = (char*)d_ws;
  _Float16* Xh = (_Float16*)ws;                 ws += BT * E * 2;            // 32 MB
  _Float16* Wh = (_Float16*)ws;                 ws += (size_t)3 * A * E * 2; //  6 MB
  _Float16* Qb = (_Float16*)ws;                 ws += BT * A * 2;            // 32 MB
  _Float16* Kb = (_Float16*)ws;                 ws += BT * A * 2;            // 32 MB
  _Float16* Vb = (_Float16*)ws;                 ws += BT * A * 2;            // 32 MB
  _Float16* Sb = (_Float16*)ws;                                             // 128 MB

  // K0: convert inputs to f16
  {
    int nX = (int)(BT * E);
    cvt_f32_to_f16<<<(nX + 255) / 256, 256, 0, stream>>>(x, Xh, nX);
    int nW = A * E;
    cvt_f32_to_f16<<<(nW + 255) / 256, 256, 0, stream>>>(Wq, Wh + 0 * (size_t)A * E, nW);
    cvt_f32_to_f16<<<(nW + 255) / 256, 256, 0, stream>>>(Wk, Wh + 1 * (size_t)A * E, nW);
    cvt_f32_to_f16<<<(nW + 255) / 256, 256, 0, stream>>>(Wv, Wh + 2 * (size_t)A * E, nW);
  }

  // K1: QKV projection
  qkv_proj<<<dim3(BT / 128, A / 64, 3), 256, 0, stream>>>(Xh, Wh, Qb, Kb, Vb);

  // K2: causal scores
  qk_scores<<<dim3(T / 128, T / 64, B), 256, 0, stream>>>(Qb, Kb, Sb);

  // K3: softmax (in place, S -> P)
  softmax_rows<<<dim3((unsigned)BT), 256, 0, stream>>>(Sb);

  // K4: output
  attn_out<<<dim3(T / 128, A / 64, B), 256, 0, stream>>>(Sb, Vb, (float*)d_out);
}
